// MistralAttention_78589311582594
// MI455X (gfx1250) — matrix operand order repaired and checked by reading, not compiled, not run on hardware
//
#include <hip/hip_runtime.h>

typedef __bf16   v16bf __attribute__((ext_vector_type(16)));
typedef float    v8f   __attribute__((ext_vector_type(8)));
typedef float    v4f   __attribute__((ext_vector_type(4)));
typedef unsigned v4u   __attribute__((ext_vector_type(4)));

__device__ __forceinline__ unsigned short f2bf(float f) {
    union { float f; unsigned u; } c; c.f = f;
    unsigned r = c.u + 0x7FFFu + ((c.u >> 16) & 1u);
    return (unsigned short)(r >> 16);
}
__device__ __forceinline__ float bf2f(unsigned short u) {
    union { float f; unsigned u; } c; c.u = ((unsigned)u) << 16;
    return c.f;
}

union Frag { v16bf v; unsigned u[8]; v4u q[2]; };

__device__ __forceinline__ v8f wmma_bf16(const Frag& a, const Frag& b, v8f c) {
    return __builtin_amdgcn_wmma_f32_16x16x32_bf16(
        /*neg_a=*/false, a.v, /*neg_b=*/false, b.v,
        /*c_mod=*/(short)0, c, /*reuse_a=*/false, /*reuse_b=*/false);
}

// ---- gfx1250 async global->LDS copy (ASYNCcnt path; not covered by builtins) ----
__device__ __forceinline__ unsigned lds_off(const void* p) {
    // flat->LDS aperture: addr[31:0] is the wave-relative LDS byte address
    return (unsigned)(unsigned long long)p;
}
__device__ __forceinline__ void async_load_b128(unsigned ldsaddr, const void* gptr) {
    asm volatile("global_load_async_to_lds_b128 %0, %1, off"
                 :: "v"(ldsaddr), "v"(gptr) : "memory");
}
__device__ __forceinline__ void wait_async() {
    asm volatile("s_wait_asynccnt 0" ::: "memory");
}

// ---------------- fp32 -> bf16 bulk convert (bandwidth-bound pre-pass) ----------------
__global__ __launch_bounds__(256) void cvt_kernel(const float* __restrict__ src,
                                                  unsigned short* __restrict__ dst, int n4) {
    int i = blockIdx.x * blockDim.x + threadIdx.x;
    if (i >= n4) return;
    v4f v = *(const v4f*)(src + (size_t)i * 4);
    union { unsigned short s[4]; unsigned long long u; } o;
    o.s[0] = f2bf(v.x); o.s[1] = f2bf(v.y); o.s[2] = f2bf(v.z); o.s[3] = f2bf(v.w);
    *(unsigned long long*)(dst + (size_t)i * 4) = o.u;
}

// ---------------- bf16 GEMM: C[M,N] = A[M,K] * B[K,N], f32 accumulate ----------------
#define BM 128
#define BN 128
#define BK 32
#define LSTR 40   // padded LDS k-stride: 80 bytes -> every row 16B-aligned for b128 ops

template<bool CF32>
__global__ __launch_bounds__(256) void gemm_wmma_kernel(const unsigned short* __restrict__ A,
                                                        const unsigned short* __restrict__ B,
                                                        void* Cp, int M, int N, int K) {
    __shared__ unsigned short As[BM * LSTR];   // [m][k]
    __shared__ unsigned short Bs[BN * LSTR];   // [n][k] (B transposed into LDS)
    const int t = threadIdx.x;
    const int m0 = blockIdx.y * BM;
    const int n0 = blockIdx.x * BN;
    const int w = t >> 5, lane = t & 31;
    const int wm = w & 3, wn = w >> 2;         // 4x2 wave grid, wave tile 32x64
    const int half = lane >> 4, l16 = lane & 15;

    v8f acc[2][4];
    for (int i = 0; i < 2; i++) for (int j = 0; j < 4; j++) acc[i][j] = {};

    for (int k0 = 0; k0 < K; k0 += BK) {
        // stage A tile 128x32: async global->LDS b128 (no transpose needed)
        for (int i = 0; i < 2; i++) {
            int idx = t + i * 256;
            int row = idx >> 2, c8 = (idx & 3) << 3;
            const unsigned short* src = A + (size_t)(m0 + row) * K + k0 + c8;
            async_load_b128(lds_off(&As[row * LSTR + c8]), src);
            if (k0 + BK < K) __builtin_prefetch(src + BK, 0, 1);
        }
        // stage B tile 32x128 transposed to [n][k] (VGPR path: transpose)
        for (int i = 0; i < 2; i++) {
            int idx = t + i * 256;
            int kr = idx >> 4, n8 = (idx & 15) << 3;
            const unsigned short* src = B + (size_t)(k0 + kr) * N + n0 + n8;
            union { v4u v; unsigned short s[8]; } bb;
            bb.v = *(const v4u*)src;
            for (int j = 0; j < 8; j++)
                Bs[(n8 + j) * LSTR + kr] = bb.s[j];
            if (k0 + BK < K) __builtin_prefetch(src + (size_t)BK * N, 0, 1);
        }
        wait_async();
        __syncthreads();

        // fragments: 2 x ds_load_b128 each (ISA bf16 A/B layouts)
        Frag a[2], b[4];
        for (int i = 0; i < 2; i++) {
            const unsigned short* p = &As[(wm * 32 + i * 16 + l16) * LSTR + half * 8];
            a[i].q[0] = *(const v4u*)p;
            a[i].q[1] = *(const v4u*)(p + 16);
        }
        for (int j = 0; j < 4; j++) {
            const unsigned short* p = &Bs[(wn * 64 + j * 16 + l16) * LSTR + half * 8];
            b[j].q[0] = *(const v4u*)p;
            b[j].q[1] = *(const v4u*)(p + 16);
        }
        for (int i = 0; i < 2; i++)
            for (int j = 0; j < 4; j++)
                acc[i][j] = wmma_bf16(a[i], b[j], acc[i][j]);
        __syncthreads();
    }

    for (int i = 0; i < 2; i++) for (int j = 0; j < 4; j++) {
        int mbase = m0 + wm * 32 + i * 16 + half * 8;
        int col = n0 + wn * 64 + j * 16 + l16;
        for (int r = 0; r < 8; r++) {
            float vv = acc[i][j][r];
            if (CF32) ((float*)Cp)[(size_t)(mbase + r) * N + col] = vv;
            else      ((unsigned short*)Cp)[(size_t)(mbase + r) * N + col] = f2bf(vv);
        }
    }
}

// ---------------- in-place RoPE on bf16 X[S, nheads, 128] ----------------
__global__ __launch_bounds__(256) void rope_kernel(unsigned short* X, int nheads, int S) {
    int idx = blockIdx.x * blockDim.x + threadIdx.x;
    int total = S * nheads * 64;
    if (idx >= total) return;
    int i = idx & 63;
    int rest = idx >> 6;
    int hh = rest % nheads;
    int p = rest / nheads;
    float inv = __expf(-(float)i * 0.14391156514261227f);  // 10000^(-i/64)
    float ang = (float)p * inv;
    float s, c;
    __sincosf(ang, &s, &c);
    size_t base = ((size_t)p * nheads + hh) * 128 + i;
    float x1 = bf2f(X[base]), x2 = bf2f(X[base + 64]);
    X[base]      = f2bf(x1 * c - x2 * s);
    X[base + 64] = f2bf(x2 * c + x1 * s);
}

// ---------------- flash attention: grid (32 heads, 16 q-blocks of 128) ----------------
#define KSTR 136   // 272B rows, 16B aligned
#define VSTR 72    // 144B rows, 16B aligned
__global__ __launch_bounds__(256) void attn_kernel(const unsigned short* __restrict__ Q,
                                                   const unsigned short* __restrict__ Kb,
                                                   const unsigned short* __restrict__ Vb,
                                                   unsigned short* __restrict__ O) {
    const int h = blockIdx.x;
    const int qblk = blockIdx.y;
    const int kvh = h >> 2;                  // GQA: 4 q-heads per kv-head
    const int t = threadIdx.x;
    const int w = t >> 5, lane = t & 31;
    const int half = lane >> 4, l16 = lane & 15;
    const int qbase = qblk * 128 + w * 16;

    __shared__ unsigned short Ks[64 * KSTR];      // [key][d]
    __shared__ unsigned short Vs[128 * VSTR];     // [d][key] (transposed)
    __shared__ unsigned short Ps[8 * 16 * VSTR];  // per-wave P tiles [row][key]

    // Q fragments in registers for whole kernel (4 k-steps of 32), b128 global loads
    Frag qa[4];
    {
        const unsigned short* qrow = Q + (size_t)(qbase + l16) * 4096 + h * 128;
        for (int st = 0; st < 4; st++) {
            const unsigned short* p = qrow + half * 8 + st * 32;
            qa[st].q[0] = *(const v4u*)p;
            qa[st].q[1] = *(const v4u*)(p + 16);
        }
    }

    v8f od[8];
    for (int i = 0; i < 8; i++) od[i] = {};
    float mrow[8], lrow[8];
    for (int r = 0; r < 8; r++) { mrow[r] = -1e30f; lrow[r] = 0.0f; }

    const float scale = 0.08838834764831845f;     // 1/sqrt(128)
    const int ntiles = qblk * 2 + 2;              // causal

    for (int tt = 0; tt < ntiles; tt++) {
        const int k0 = tt * 64;
        __syncthreads();
        // stage K [64x128] via async b128; V transposed [128x64] via VGPR path
        for (int i = 0; i < 4; i++) {
            int idx = t + i * 256;                 // 1024 chunks of 8 bf16
            int row = idx >> 4, d8 = (idx & 15) << 3;
            const size_t goff = (size_t)(k0 + row) * 1024 + kvh * 128 + d8;
            async_load_b128(lds_off(&Ks[row * KSTR + d8]), Kb + goff);
            union { v4u v; unsigned short s[8]; } vv;
            vv.v = *(const v4u*)(Vb + goff);
            for (int j = 0; j < 8; j++)
                Vs[(d8 + j) * VSTR + row] = vv.s[j];
            if (tt + 1 < ntiles) {
                __builtin_prefetch(Kb + goff + 64 * 1024, 0, 1);
                __builtin_prefetch(Vb + goff + 64 * 1024, 0, 1);
            }
        }
        wait_async();
        __syncthreads();

        // S = Q K^T : 4 key-tiles x 4 d-steps
        v8f sc[4];
        for (int j = 0; j < 4; j++) {
            v8f s = {};
            int key = j * 16 + l16;
            for (int st = 0; st < 4; st++) {
                Frag b;
                const unsigned short* p = &Ks[key * KSTR + half * 8 + st * 32];
                b.q[0] = *(const v4u*)p;
                b.q[1] = *(const v4u*)(p + 16);
                s = wmma_bf16(qa[st], b, s);
            }
            sc[j] = s;
        }
        // scale + causal mask (reference adds -1e9)
        for (int j = 0; j < 4; j++) {
            int key = k0 + j * 16 + l16;
            for (int r = 0; r < 8; r++) {
                int q = qbase + r + half * 8;
                float v = sc[j][r] * scale;
                if (key > q) v += -1.0e9f;
                sc[j][r] = v;
            }
        }
        // online softmax (row stats via intra-16-lane butterflies)
        float mnew[8], psum[8];
        for (int r = 0; r < 8; r++) {
            float mx = fmaxf(fmaxf(sc[0][r], sc[1][r]), fmaxf(sc[2][r], sc[3][r]));
            for (int d = 1; d < 16; d <<= 1)
                mx = fmaxf(mx, __shfl_xor(mx, d, 32));
            mnew[r] = fmaxf(mrow[r], mx);
        }
        for (int r = 0; r < 8; r++) {
            float sum = 0.f;
            for (int j = 0; j < 4; j++) {
                float p = __expf(sc[j][r] - mnew[r]);
                sc[j][r] = p;
                sum += p;
            }
            for (int d = 1; d < 16; d <<= 1)
                sum += __shfl_xor(sum, d, 32);
            psum[r] = sum;
        }
        for (int r = 0; r < 8; r++) {
            float corr = __expf(mrow[r] - mnew[r]);
            lrow[r] = lrow[r] * corr + psum[r];
            mrow[r] = mnew[r];
            for (int dt = 0; dt < 8; dt++)
                od[dt][r] *= corr;
        }
        // stage P bf16 (C-layout -> A-layout via LDS)
        unsigned short* pw = &Ps[w * 16 * VSTR];
        for (int j = 0; j < 4; j++)
            for (int r = 0; r < 8; r++)
                pw[(r + half * 8) * VSTR + j * 16 + l16] = f2bf(sc[j][r]);
        __syncthreads();
        // O += P V : 2 key-steps x 8 d-tiles
        for (int ks = 0; ks < 2; ks++) {
            Frag pa;
            const unsigned short* pp = &pw[l16 * VSTR + half * 8 + ks * 32];
            pa.q[0] = *(const v4u*)pp;
            pa.q[1] = *(const v4u*)(pp + 16);
            for (int dt = 0; dt < 8; dt++) {
                Frag b;
                const unsigned short* p = &Vs[(dt * 16 + l16) * VSTR + half * 8 + ks * 32];
                b.q[0] = *(const v4u*)p;
                b.q[1] = *(const v4u*)(p + 16);
                od[dt] = wmma_bf16(pa, b, od[dt]);
            }
        }
    }

    // normalize and write attn output (bf16, [S, NH*HD])
    for (int dt = 0; dt < 8; dt++) {
        int col = h * 128 + dt * 16 + l16;
        for (int r = 0; r < 8; r++) {
            int row = qbase + r + half * 8;
            O[(size_t)row * 4096 + col] = f2bf(od[dt][r] / lrow[r]);
        }
    }
}

extern "C" void kernel_launch(void* const* d_in, const int* in_sizes, int n_in,
                              void* d_out, int out_size, void* d_ws, size_t ws_size,
                              hipStream_t stream) {
    const float* hs = (const float*)d_in[0];
    const float* Wq = (const float*)d_in[3];
    const float* Wk = (const float*)d_in[4];
    const float* Wv = (const float*)d_in[5];
    const float* Wo = (const float*)d_in[6];
    float* out = (float*)d_out;

    const int S = 2048, H = 4096, KVD = 1024;

    // workspace layout (all bf16)
    unsigned short* HSb = (unsigned short*)d_ws;             // [2048,4096]
    unsigned short* Wqb = HSb + (size_t)S * H;               // [4096,4096]
    unsigned short* Wkb = Wqb + (size_t)H * H;               // [4096,1024]
    unsigned short* Wvb = Wkb + (size_t)H * KVD;             // [4096,1024]
    unsigned short* Wob = Wvb + (size_t)H * KVD;             // [4096,4096]
    unsigned short* Qb  = Wob + (size_t)H * H;               // [2048,4096]
    unsigned short* Kb  = Qb  + (size_t)S * H;               // [2048,1024]
    unsigned short* Vb  = Kb  + (size_t)S * KVD;             // [2048,1024]
    unsigned short* AOb = Vb  + (size_t)S * KVD;             // [2048,4096]

    dim3 blk(256);
    // fp32 -> bf16 pre-pass (removes all conversion work from GEMM hot loops)
    {
        int n;
        n = (S * H) / 4;   cvt_kernel<<<(n + 255) / 256, blk, 0, stream>>>(hs, HSb, n);
        n = (H * H) / 4;   cvt_kernel<<<(n + 255) / 256, blk, 0, stream>>>(Wq, Wqb, n);
        n = (H * KVD) / 4; cvt_kernel<<<(n + 255) / 256, blk, 0, stream>>>(Wk, Wkb, n);
        n = (H * KVD) / 4; cvt_kernel<<<(n + 255) / 256, blk, 0, stream>>>(Wv, Wvb, n);
        n = (H * H) / 4;   cvt_kernel<<<(n + 255) / 256, blk, 0, stream>>>(Wo, Wob, n);
    }
    // QKV projections
    gemm_wmma_kernel<false><<<dim3(H / BN, S / BM), blk, 0, stream>>>(HSb, Wqb, Qb, S, H, H);
    gemm_wmma_kernel<false><<<dim3(KVD / BN, S / BM), blk, 0, stream>>>(HSb, Wkb, Kb, S, KVD, H);
    gemm_wmma_kernel<false><<<dim3(KVD / BN, S / BM), blk, 0, stream>>>(HSb, Wvb, Vb, S, KVD, H);
    // RoPE on Q and K
    int nq = S * 32 * 64;
    rope_kernel<<<(nq + 255) / 256, blk, 0, stream>>>(Qb, 32, S);
    int nk = S * 8 * 64;
    rope_kernel<<<(nk + 255) / 256, blk, 0, stream>>>(Kb, 8, S);
    // Flash attention
    attn_kernel<<<dim3(32, S / 128), blk, 0, stream>>>(Qb, Kb, Vb, AOb);
    // Output projection (fp32 result)
    gemm_wmma_kernel<true><<<dim3(H / BN, S / BM), blk, 0, stream>>>(AOb, Wob, out, S, H, H);
}
